// GaussianMixture_45835890983250
// MI455X (gfx1250) — matrix operand order repaired and checked by reading, not compiled, not run on hardware
//
#include <hip/hip_runtime.h>

#define NROWS 16384
#define CDIM  512
#define HDIM  1024
#define FDIM  512
#define KEXP  5

#define LDS_STRIDE 132
#define LDS_TILE   (16 * LDS_STRIDE)

typedef __attribute__((ext_vector_type(16))) __bf16 v16bf;
typedef __attribute__((ext_vector_type(8)))  float  v8f;
typedef int gvec4i __attribute__((vector_size(16)));   // matches builtin param type

union Frag { unsigned int u[8]; v16bf v; };

// gfx1250 async global->LDS DMA path (ASYNCcnt-tracked), with sync fallback.
#if defined(__gfx1250__) && \
    __has_builtin(__builtin_amdgcn_global_load_async_to_lds_b32) && \
    __has_builtin(__builtin_amdgcn_global_load_async_to_lds_b128) && \
    __has_builtin(__builtin_amdgcn_s_wait_asynccnt)
#define GEMM_ASYNC 1
#else
#define GEMM_ASYNC 0
#endif

// typed address-space pointer casts (value-preserving integer casts: AS1 value
// == generic value; LDS offset lives in the low 32 bits of a generic pointer)
#define G_I32(p) ((__attribute__((address_space(1))) int*)(unsigned long long)(p))
#define G_V4I(p) ((__attribute__((address_space(1))) gvec4i*)(unsigned long long)(p))
#define L_I32(p) ((__attribute__((address_space(3))) int*)(unsigned int)(unsigned long long)(p))
#define L_V4I(p) ((__attribute__((address_space(3))) gvec4i*)(unsigned int)(unsigned long long)(p))

__device__ __forceinline__ unsigned short f2bf(float f) {
  unsigned int x = __float_as_uint(f);
  unsigned int r = x + 0x7FFFu + ((x >> 16) & 1u);   // round-to-nearest-even
  return (unsigned short)(r >> 16);
}
__device__ __forceinline__ float bf2f(unsigned short h) {
  return __uint_as_float(((unsigned int)h) << 16);
}

// ---------------- f32 -> bf16 (activations) ----------------
__global__ __launch_bounds__(256) void cvt_bf16_kernel(const float* __restrict__ in,
                                                       unsigned short* __restrict__ out,
                                                       int n) {
  int i = blockIdx.x * 256 + threadIdx.x;
  if (i < n) out[i] = f2bf(in[i]);
}

// ---------------- f32 -> packed-pair bf16 (weights) ----------------
// out[k2*Nc + n] = { bf16(in[(2k2)*Nc+n]), bf16(in[(2k2+1)*Nc+n]) }
// i.e. B pre-interleaved into the WMMA B-fragment dword layout, so GEMM
// staging is a straight b128 copy with zero in-loop packing.
__global__ __launch_bounds__(256) void cvt_pack_kernel(const float* __restrict__ in,
                                                       unsigned int* __restrict__ out,
                                                       int Nc) {
  int n  = blockIdx.x * 256 + threadIdx.x;
  int k2 = blockIdx.y;
  size_t base = (size_t)(2 * k2) * Nc + n;
  unsigned int lo = f2bf(in[base]);
  unsigned int hi = f2bf(in[base + Nc]);
  out[(size_t)k2 * Nc + n] = lo | (hi << 16);
}

// ---------------- Tiled WMMA GEMM: D = act(A @ B + bias) ----------------
// A : M x Kd bf16 row-major (activations).
// Bp: (Kd/2) x Nc uint32 K-pair-packed bf16 (weights).
// Block tile 128x128, BK=32, double-buffered LDS, async global->LDS staging.
// 8 waves; each wave owns a 32x64 sub-tile (2x4 of 16x16 WMMA accumulators).
__global__ __launch_bounds__(256) void gemm_bf16_wmma(
    const unsigned short* __restrict__ A, const unsigned int* __restrict__ Bp,
    const float* __restrict__ bias, unsigned short* __restrict__ Cb,
    float* __restrict__ Cf, int M, int Nc, int Kd, int relu)
{
  __shared__ unsigned int ldsA[2 * LDS_TILE];  // [buf][k2][m], stride 132
  __shared__ unsigned int ldsB[2 * LDS_TILE];  // [buf][k2][n], stride 132

  const int tid  = threadIdx.x;
  const int lane = tid & 31;
  const int wave = tid >> 5;
  const int rowBase = (wave >> 1) * 32;   // 0,32,64,96
  const int colBase = (wave & 1) * 64;    // 0,64

  const int bm = blockIdx.y * 128;
  const int bn = blockIdx.x * 128;

  v8f acc[2][4];
  v8f vzero = {0.f, 0.f, 0.f, 0.f, 0.f, 0.f, 0.f, 0.f};
#pragma unroll
  for (int s = 0; s < 2; ++s)
#pragma unroll
    for (int t = 0; t < 4; ++t) acc[s][t] = vzero;

  const int lan15 = lane & 15;
  const int lhi   = lane >> 4;
  const int k2bA  = lhi * 8;   // lanes 16-31 read K=16..31
  const int k2bB  = lhi * 8;   // lanes 16-31 read K=16..31

  // ---- tile staging (global -> LDS), buf in {0,1} ----
  auto stage = [&](int buf, int k0) {
#if GEMM_ASYNC
    // A tile: 128 rows x 16 pair-dwords. One b32 per lane; lanes 0-15 cover
    // row r (64B contiguous), lanes 16-31 row r+1.
#pragma unroll
    for (int i = 0; i < 8; ++i) {
      int row = wave * 16 + i * 2 + lhi;
      const unsigned short* ga = A + (size_t)(bm + row) * Kd + k0 + 2 * lan15;
      unsigned int* la = &ldsA[buf * LDS_TILE + lan15 * LDS_STRIDE + row];
      __builtin_amdgcn_global_load_async_to_lds_b32(G_I32(ga), L_I32(la), 0, 0);
    }
    // B tile: 16 packed rows x 128 dwords. One b128 per lane; a wave covers
    // one full 512B packed row per instruction.
#pragma unroll
    for (int i = 0; i < 2; ++i) {
      int k2 = wave * 2 + i;
      const unsigned int* gb = Bp + (size_t)(k0 / 2 + k2) * Nc + bn + lane * 4;
      unsigned int* lb = &ldsB[buf * LDS_TILE + k2 * LDS_STRIDE + lane * 4];
      __builtin_amdgcn_global_load_async_to_lds_b128(G_V4I(gb), L_V4I(lb), 0, 0);
    }
#else
    // sync fallback: b128 loads + ds stores (no conditional prefetch)
#pragma unroll
    for (int it = 0; it < 2; ++it) {
      int task  = tid + it * 256;        // 0..511
      int row   = task >> 2;             // 0..127
      int chunk = task & 3;              // 8 bf16 = 4 pair-dwords
      const unsigned short* pa = A + (size_t)(bm + row) * Kd + k0 + chunk * 8;
      uint4 d = *reinterpret_cast<const uint4*>(pa);
      int k2 = chunk * 4;
      unsigned int* la = &ldsA[buf * LDS_TILE];
      la[(k2 + 0) * LDS_STRIDE + row] = d.x;
      la[(k2 + 1) * LDS_STRIDE + row] = d.y;
      la[(k2 + 2) * LDS_STRIDE + row] = d.z;
      la[(k2 + 3) * LDS_STRIDE + row] = d.w;
    }
#pragma unroll
    for (int it = 0; it < 2; ++it) {
      int task = tid + it * 256;         // 0..511, 4 dwords each
      int k2   = task >> 5;
      int cch  = task & 31;
      const uint4* src = reinterpret_cast<const uint4*>(
          Bp + (size_t)(k0 / 2 + k2) * Nc + bn + cch * 4);
      *reinterpret_cast<uint4*>(&ldsB[buf * LDS_TILE + k2 * LDS_STRIDE + cch * 4]) = *src;
    }
#endif
  };

  const int nk = Kd >> 5;
  stage(0, 0);
#if GEMM_ASYNC
  __builtin_amdgcn_s_wait_asynccnt(0);
#endif
  __syncthreads();

  for (int kt = 0; kt < nk; ++kt) {
    const int cur = kt & 1;
    if (kt + 1 < nk) stage(cur ^ 1, (kt + 1) << 5);   // DMA next tile during compute

    // ---- build fragments from current buffer, issue 8 WMMAs ----
    const unsigned int* la = &ldsA[cur * LDS_TILE];
    const unsigned int* lb = &ldsB[cur * LDS_TILE];
    Frag afr[2], bfr[4];
#pragma unroll
    for (int s = 0; s < 2; ++s) {
      int m = rowBase + s * 16 + lan15;
#pragma unroll
      for (int v = 0; v < 4; ++v) afr[s].u[v]     = la[(k2bA + v) * LDS_STRIDE + m];
#pragma unroll
      for (int v = 0; v < 4; ++v) afr[s].u[4 + v] = la[(k2bA + 4 + v) * LDS_STRIDE + m];
    }
#pragma unroll
    for (int t = 0; t < 4; ++t) {
      int n = colBase + t * 16 + lan15;
#pragma unroll
      for (int v = 0; v < 8; ++v) bfr[t].u[v] = lb[(k2bB + v) * LDS_STRIDE + n];
    }
#pragma unroll
    for (int s = 0; s < 2; ++s)
#pragma unroll
      for (int t = 0; t < 4; ++t)
        acc[s][t] = __builtin_amdgcn_wmma_f32_16x16x32_bf16(
            false, afr[s].v, false, bfr[t].v, (short)0, acc[s][t], false, false);

#if GEMM_ASYNC
    __builtin_amdgcn_s_wait_asynccnt(0);
#endif
    __syncthreads();
  }

  // ---- epilogue: bias + optional ReLU, store bf16 or f32 ----
  const int rOff = lhi * 8;   // C/D layout: lanes 16-31 hold M=r+8
#pragma unroll
  for (int s = 0; s < 2; ++s) {
#pragma unroll
    for (int t = 0; t < 4; ++t) {
      int gc = bn + colBase + t * 16 + lan15;
      float bv = bias[gc];
#pragma unroll
      for (int r = 0; r < 8; ++r) {
        int gr = bm + rowBase + s * 16 + rOff + r;
        float v = acc[s][t][r] + bv;
        if (relu) v = v > 0.f ? v : 0.f;
        if (Cb) Cb[(size_t)gr * Nc + gc] = f2bf(v);
        else    Cf[(size_t)gr * Nc + gc] = v;
      }
    }
  }
}

// ---------------- router head: logits -> softmax -> gumbel argmax -> weights ----------------
__global__ __launch_bounds__(256) void head_kernel(const unsigned short* __restrict__ h2,
                                                   const float* __restrict__ pw2,
                                                   const float* __restrict__ pb2,
                                                   const float* __restrict__ gum,
                                                   float* __restrict__ wts)
{
  const int lane = threadIdx.x & 31;
  const int wave = threadIdx.x >> 5;
  const int n = blockIdx.x * 8 + wave;   // one wave32 per row

  const unsigned short* hrow = h2 + (size_t)n * HDIM;
  float a0 = 0.f, a1 = 0.f, a2 = 0.f, a3 = 0.f, a4 = 0.f;
  for (int kk = lane; kk < HDIM; kk += 32) {
    float hv = bf2f(hrow[kk]);
    const float* w = pw2 + (size_t)kk * KEXP;
    a0 += hv * w[0]; a1 += hv * w[1]; a2 += hv * w[2]; a3 += hv * w[3]; a4 += hv * w[4];
  }
#pragma unroll
  for (int o = 16; o > 0; o >>= 1) {
    a0 += __shfl_xor(a0, o, 32);
    a1 += __shfl_xor(a1, o, 32);
    a2 += __shfl_xor(a2, o, 32);
    a3 += __shfl_xor(a3, o, 32);
    a4 += __shfl_xor(a4, o, 32);
  }
  if (lane == 0) {
    const float EPS = 1e-20f, WEIGHT = 5.0f;
    float lg[KEXP] = {a0 + pb2[0], a1 + pb2[1], a2 + pb2[2], a3 + pb2[3], a4 + pb2[4]};
    float mx = lg[0];
#pragma unroll
    for (int j = 1; j < KEXP; ++j) mx = lg[j] > mx ? lg[j] : mx;
    float ex[KEXP], sum = 0.f;
#pragma unroll
    for (int j = 0; j < KEXP; ++j) { ex[j] = expf(lg[j] - mx); sum += ex[j]; }
    float inv = 1.f / sum;
    int idx = 0; float best = -3.4e38f;
#pragma unroll
    for (int j = 0; j < KEXP; ++j) {
      float ps = ex[j] * inv;
      float u  = gum[(size_t)n * KEXP + j];
      float gb = -logf(-logf(u + EPS) + EPS);
      float sc = logf(ps + EPS) + gb;
      if (sc > best) { best = sc; idx = j; }
    }
#pragma unroll
    for (int j = 0; j < KEXP; ++j) {
      float ps = ex[j] * inv;
      wts[(size_t)n * KEXP + j] = (ps + (j == idx ? WEIGHT : 0.f)) / (1.f + WEIGHT);
    }
  }
}

// ---------------- reparameterize + weighted mixture accumulate ----------------
__global__ __launch_bounds__(256) void epilogue_kernel(const float* __restrict__ o,
                                                       const float* __restrict__ nz,
                                                       const float* __restrict__ wts,
                                                       float* __restrict__ out, int k)
{
  size_t i = (size_t)blockIdx.x * 256 + threadIdx.x;   // over N*FDIM
  size_t n = i >> 9;         // FDIM == 512
  size_t f = i & 511;
  float mean = o[n * (size_t)(2 * FDIM) + f];
  float logv = o[n * (size_t)(2 * FDIM) + FDIM + f];
  float s = mean + nz[i] * expf(0.5f * logv);
  float v = wts[n * KEXP + k] * s;
  if (k == 0) out[i] = v;
  else        out[i] += v;
}

extern "C" void kernel_launch(void* const* d_in, const int* in_sizes, int n_in,
                              void* d_out, int out_size, void* d_ws, size_t ws_size,
                              hipStream_t stream)
{
  const float* c     = (const float*)d_in[0];
  // d_in[1] = x (N,3): unused by the reference
  const float* noise = (const float*)d_in[2];
  const float* gum   = (const float*)d_in[3];
  const float* pw0   = (const float*)d_in[4];
  const float* pb0   = (const float*)d_in[5];
  const float* pw1   = (const float*)d_in[6];
  const float* pb1   = (const float*)d_in[7];
  const float* pw2   = (const float*)d_in[8];
  const float* pb2   = (const float*)d_in[9];
  const float* gw0   = (const float*)d_in[10];
  const float* gb0   = (const float*)d_in[11];
  const float* gw1   = (const float*)d_in[12];
  const float* gb1   = (const float*)d_in[13];
  const float* gw2   = (const float*)d_in[14];
  const float* gb2   = (const float*)d_in[15];
  float* out = (float*)d_out;

  char* ws = (char*)d_ws;
  size_t off = 0;
  auto take = [&](size_t bytes) -> char* {
    char* p = ws + off;
    off += (bytes + 255) & ~(size_t)255;
    return p;
  };
  unsigned short* c_bf  = (unsigned short*)take((size_t)NROWS * CDIM * 2);
  unsigned short* h1    = (unsigned short*)take((size_t)NROWS * HDIM * 2);   // reused as expert g1
  unsigned short* h2    = (unsigned short*)take((size_t)NROWS * HDIM * 2);   // reused as expert g2
  unsigned int*   pw0_p = (unsigned int*)take((size_t)(CDIM / 2) * HDIM * 4);
  unsigned int*   pw1_p = (unsigned int*)take((size_t)(HDIM / 2) * HDIM * 4);
  unsigned int*   gw0_p = (unsigned int*)take((size_t)KEXP * (CDIM / 2) * HDIM * 4);
  unsigned int*   gw1_p = (unsigned int*)take((size_t)KEXP * (HDIM / 2) * HDIM * 4);
  unsigned int*   gw2_p = (unsigned int*)take((size_t)KEXP * (HDIM / 2) * (2 * FDIM) * 4);
  float* wts  = (float*)take((size_t)NROWS * KEXP * 4);
  float* obuf = (float*)take((size_t)NROWS * (2 * FDIM) * 4);

  // activations -> bf16
  {
    size_t n = (size_t)NROWS * CDIM;
    cvt_bf16_kernel<<<dim3((unsigned)((n + 255) / 256)), dim3(256), 0, stream>>>(c, c_bf, (int)n);
  }
  // weights -> K-pair-packed bf16 (expert row counts are even, so pairs never
  // straddle an expert boundary; one launch per stacked weight tensor)
  auto cvtp = [&](const float* src, unsigned int* dst, int rows, int cols) {
    cvt_pack_kernel<<<dim3(cols / 256, rows / 2), dim3(256), 0, stream>>>(src, dst, cols);
  };
  cvtp(pw0, pw0_p, CDIM, HDIM);
  cvtp(pw1, pw1_p, HDIM, HDIM);
  cvtp(gw0, gw0_p, KEXP * CDIM, HDIM);
  cvtp(gw1, gw1_p, KEXP * HDIM, HDIM);
  cvtp(gw2, gw2_p, KEXP * HDIM, 2 * FDIM);

  dim3 gblk(256);
  dim3 ggrid(HDIM / 128, NROWS / 128);

  // router
  gemm_bf16_wmma<<<ggrid, gblk, 0, stream>>>(c_bf, pw0_p, pb0, h1, nullptr, NROWS, HDIM, CDIM, 1);
  gemm_bf16_wmma<<<ggrid, gblk, 0, stream>>>(h1, pw1_p, pb1, h2, nullptr, NROWS, HDIM, HDIM, 1);
  head_kernel<<<dim3(NROWS / 8), dim3(256), 0, stream>>>(h2, pw2, pb2, gum, wts);

  // experts
  for (int k = 0; k < KEXP; ++k) {
    gemm_bf16_wmma<<<ggrid, gblk, 0, stream>>>(c_bf, gw0_p + (size_t)k * (CDIM / 2) * HDIM,
                                               gb0 + (size_t)k * HDIM, h1, nullptr,
                                               NROWS, HDIM, CDIM, 1);
    gemm_bf16_wmma<<<ggrid, gblk, 0, stream>>>(h1, gw1_p + (size_t)k * (HDIM / 2) * HDIM,
                                               gb1 + (size_t)k * HDIM, h2, nullptr,
                                               NROWS, HDIM, HDIM, 1);
    gemm_bf16_wmma<<<ggrid, gblk, 0, stream>>>(h2, gw2_p + (size_t)k * (HDIM / 2) * (2 * FDIM),
                                               gb2 + (size_t)k * 2 * FDIM, nullptr, obuf,
                                               NROWS, 2 * FDIM, HDIM, 0);
    epilogue_kernel<<<dim3((NROWS * FDIM) / 256), dim3(256), 0, stream>>>(
        obuf, noise + (size_t)k * NROWS * FDIM, wts, out, k);
  }
}
